// QWENWrapper_63402307224124
// MI455X (gfx1250) — matrix operand order repaired and checked by reading, not compiled, not run on hardware
//
#include <hip/hip_runtime.h>
#include <hip/hip_bf16.h>

// ---------------- model constants (from reference) ----------------
#define LAYERS 4
#define DMODEL 2048
#define NH     16
#define NKVH   8
#define DHEAD  128
#define FF     6144
#define VOCAB  32000
#define SMAXC  2048
#define TSEQ   1024

typedef __bf16 bf16_t;
typedef bf16_t v16bf __attribute__((ext_vector_type(16)));
typedef bf16_t v2bf  __attribute__((ext_vector_type(2)));
typedef float  v8f   __attribute__((ext_vector_type(8)));
typedef unsigned int v4u __attribute__((ext_vector_type(4)));
typedef int v8i __attribute__((ext_vector_type(8)));
typedef int v4i __attribute__((ext_vector_type(4)));

__device__ __forceinline__ unsigned short f32_to_bf16_rne(float f) {
  union { float f; unsigned int u; } c; c.f = f;
  unsigned int u = c.u;
  u += 0x7FFFu + ((u >> 16) & 1u);   // round-to-nearest-even
  return (unsigned short)(u >> 16);
}

// pack two f32 -> packed bf16x2 (HW packed convert when available)
__device__ __forceinline__ unsigned int pack2_bf16(float x, float y) {
#if __has_builtin(__builtin_amdgcn_cvt_pk_bf16_f32)
  union { v2bf v; unsigned int u; } c;
  c.v = __builtin_amdgcn_cvt_pk_bf16_f32(x, y);
  return c.u;
#else
  return (unsigned)f32_to_bf16_rne(x) | ((unsigned)f32_to_bf16_rne(y) << 16);
#endif
}

__device__ __forceinline__ float block_reduce_sum_256(float v, float* red) {
  int tid = threadIdx.x;
  red[tid] = v; __syncthreads();
  for (int s = 128; s > 0; s >>= 1) {
    if (tid < s) red[tid] += red[tid + s];
    __syncthreads();
  }
  float r = red[0]; __syncthreads();
  return r;
}

// ---------------------------------------------------------------------------
// Generic batched GEMM, C[M,N] (+)= A[M,K] * B[K,N], fp32 in/out, bf16 WMMA.
// Block = 256 threads (8 wave32). Block tile 128x128, BK=32.
// Wave computes 32x64: 2 A frags x 4 B frags -> 8 back-to-back WMMAs / k-step.
// Double-buffered LDS: next tile's global loads issue before the WMMAs,
// conversion+stores go to the other buffer -> one barrier per k-step.
// zdivB: kv-head sharing for GQA (B batch index = z >> zdivB).
// Requires: M%128==0, N%128==0, K%32==0, A/B 16B-aligned, lda/ldb%4==0.
// ---------------------------------------------------------------------------
__global__ __launch_bounds__(256)
void gemm_bf16_wmma(const float* __restrict__ A, int lda, long long sA,
                    const float* __restrict__ B, int ldb, long long sB, int zdivB,
                    float* __restrict__ C, int ldc, long long sC,
                    int M, int N, int K, int addC)
{
  // row stride 40 halves = 80 bytes (16B multiple -> aligned b128/b64 LDS ops)
  __shared__ __align__(16) unsigned short As[2][128][40];  // [buf][m][k]
  __shared__ __align__(16) unsigned short Bs[2][128][40];  // [buf][n][k] (transposed)

  const int tid = threadIdx.x;
  const int z   = blockIdx.z;
  A += (long long)z * sA;
  B += (long long)(z >> zdivB) * sB;
  C += (long long)z * sC;

  const int n0 = blockIdx.x * 128;
  const int m0 = blockIdx.y * 128;

  const int wave = tid >> 5;
  const int lane = tid & 31;
  const int half = lane >> 4;    // 0..1
  const int l15  = lane & 15;
  const int mw   = wave & 3;     // 4 wave-rows of 32
  const int nw   = wave >> 2;    // 2 wave-cols of 64

  // per-thread staging coordinates (fixed across k-steps)
  int rowA[4], colA[4], krB[4], ncB[4];
  const float* aPtr[4];
  const float* bPtr[4];
#pragma unroll
  for (int r = 0; r < 4; ++r) {
    int f4 = tid + r * 256;            // 1024 float4 per tile
    rowA[r] = f4 >> 3;                 // A: 8 float4 per 32-wide row
    colA[r] = (f4 & 7) * 4;
    krB[r]  = f4 >> 5;                 // B: 32 float4 per 128-wide row
    ncB[r]  = (f4 & 31) * 4;
    aPtr[r] = A + (long long)(m0 + rowA[r]) * lda + colA[r];
    bPtr[r] = B + (long long)krB[r] * ldb + (n0 + ncB[r]);
  }

  float4 av[4], bv[4];
  auto store_tile = [&](int bi) {
#pragma unroll
    for (int r = 0; r < 4; ++r) {
      uint2 p;
      p.x = pack2_bf16(av[r].x, av[r].y);
      p.y = pack2_bf16(av[r].z, av[r].w);
      *reinterpret_cast<uint2*>(&As[bi][rowA[r]][colA[r]]) = p;
      unsigned q0 = pack2_bf16(bv[r].x, bv[r].y);
      unsigned q1 = pack2_bf16(bv[r].z, bv[r].w);
      Bs[bi][ncB[r] + 0][krB[r]] = (unsigned short)q0;
      Bs[bi][ncB[r] + 1][krB[r]] = (unsigned short)(q0 >> 16);
      Bs[bi][ncB[r] + 2][krB[r]] = (unsigned short)q1;
      Bs[bi][ncB[r] + 3][krB[r]] = (unsigned short)(q1 >> 16);
    }
  };

  // prologue: load + stage tile 0
#pragma unroll
  for (int r = 0; r < 4; ++r) {
    av[r] = *reinterpret_cast<const float4*>(aPtr[r]);
    bv[r] = *reinterpret_cast<const float4*>(bPtr[r]);
  }
  store_tile(0);
  __syncthreads();

  v8f acc[2][4] = {};
  union Frag { v16bf v; uint4 q[2]; };
  int cur = 0;

  for (int k0 = 0; k0 < K; k0 += 32) {
    const bool more = (k0 + 32) < K;
    if (more) {
      // issue next tile's global loads before compute (latency hidden by WMMAs)
#pragma unroll
      for (int r = 0; r < 4; ++r) {
        av[r] = *reinterpret_cast<const float4*>(aPtr[r] + (k0 + 32));
        bv[r] = *reinterpret_cast<const float4*>(bPtr[r] + (long long)(k0 + 32) * ldb);
      }
      if (k0 + 64 < K)
        __builtin_prefetch(bPtr[0] + (long long)(k0 + 64) * ldb, 0, 1);  // global_prefetch_b8
    }

    // ---- fragments from buf[cur], then 8 WMMAs back-to-back ----
    Frag a[2], b[4];
    const int kh = half * 8;    // A: K = kh..kh+7 and kh+16..kh+23
    const int kb = half * 8;    // B: K = kb..kb+7 and kb+16..kb+23
#pragma unroll
    for (int mi = 0; mi < 2; ++mi) {
      int arow = mw * 32 + mi * 16 + l15;
      a[mi].q[0] = *reinterpret_cast<const uint4*>(&As[cur][arow][kh]);
      a[mi].q[1] = *reinterpret_cast<const uint4*>(&As[cur][arow][kh + 16]);
    }
#pragma unroll
    for (int j = 0; j < 4; ++j) {
      int brow = nw * 64 + j * 16 + l15;
      b[j].q[0] = *reinterpret_cast<const uint4*>(&Bs[cur][brow][kb]);
      b[j].q[1] = *reinterpret_cast<const uint4*>(&Bs[cur][brow][kb + 16]);
    }
#pragma unroll
    for (int mi = 0; mi < 2; ++mi)
#pragma unroll
      for (int j = 0; j < 4; ++j)
        acc[mi][j] = __builtin_amdgcn_wmma_f32_16x16x32_bf16(
            false, a[mi].v, false, b[j].v, (short)0, acc[mi][j], false, false);

    if (more) {
      store_tile(cur ^ 1);   // fill the other buffer
      __syncthreads();       // single barrier per k-step
      cur ^= 1;
    }
  }

  // C/D layout: VGPR i -> M = i + half*8, N = l15
#pragma unroll
  for (int mi = 0; mi < 2; ++mi) {
#pragma unroll
    for (int j = 0; j < 4; ++j) {
      int ccol = n0 + nw * 64 + j * 16 + l15;
#pragma unroll
      for (int i = 0; i < 8; ++i) {
        int crow = m0 + mw * 32 + mi * 16 + half * 8 + i;
        long long off = (long long)crow * ldc + ccol;
        float v = acc[mi][j][i];
        if (addC) v += C[off];
        C[off] = v;
      }
    }
  }
}

// ---------------- embedding dequant ----------------
__global__ __launch_bounds__(256)
void embed_kernel(const int* __restrict__ ids, const unsigned char* __restrict__ data,
                  const float* __restrict__ scale, const float* __restrict__ zero,
                  float* __restrict__ hidden)
{
  int i = blockIdx.x * 256 + threadIdx.x;
  if (i >= TSEQ * DMODEL) return;
  int t = i / DMODEL, d = i - t * DMODEL;
  int id = ids[t];
  hidden[i] = (float)data[(long long)id * DMODEL + d] * scale[id] + zero[id];
}

// ---------------- RMSNorm over rows ----------------
__global__ __launch_bounds__(256)
void rmsnorm_kernel(const float* __restrict__ src, float* __restrict__ dst,
                    const float* __restrict__ w, int cols)
{
  __shared__ float red[256];
  const int row = blockIdx.x;
  const float* x = src + (long long)row * cols;
  float ss = 0.f;
  for (int c = threadIdx.x; c < cols; c += 256) { float v = x[c]; ss += v * v; }
  ss = block_reduce_sum_256(ss, red);
  float inv = rsqrtf(ss / cols + 1e-6f);
  float* y = dst + (long long)row * cols;
  for (int c = threadIdx.x; c < cols; c += 256) y[c] = w[c] * (x[c] * inv);
}

// ---------------- per-head RMS + RoPE for Q -> (H,T,DH) ----------------
__global__ __launch_bounds__(128)
void q_rope_kernel(const float* __restrict__ q, float* __restrict__ qrot,
                   const float* __restrict__ wq, const float* __restrict__ cosE,
                   const float* __restrict__ sinE)
{
  __shared__ float red[128];
  __shared__ float xn[128];
  int h = blockIdx.x, t = blockIdx.y, d = threadIdx.x;
  float x = q[((long long)t * NH + h) * DHEAD + d];
  red[d] = x * x; __syncthreads();
  for (int s = 64; s > 0; s >>= 1) { if (d < s) red[d] += red[d + s]; __syncthreads(); }
  float inv = rsqrtf(red[0] / DHEAD + 1e-6f);
  float v = wq[d] * (x * inv);
  xn[d] = v; __syncthreads();
  float rot = (d < 64) ? -xn[d + 64] : xn[d - 64];
  float out = v * cosE[t * DHEAD + d] + rot * sinE[t * DHEAD + d];
  qrot[((long long)h * TSEQ + t) * DHEAD + d] = out;
}

// ------- per-head RMS + RoPE for K -> key cache [h][d][s] (out + aligned ws) -------
__global__ __launch_bounds__(128)
void k_rope_kernel(const float* __restrict__ k, float* __restrict__ kOutL,
                   float* __restrict__ kc, const float* __restrict__ wk,
                   const float* __restrict__ cosE, const float* __restrict__ sinE,
                   int past)
{
  __shared__ float red[128];
  __shared__ float xn[128];
  int h = blockIdx.x, t = blockIdx.y, d = threadIdx.x;
  float x = k[((long long)t * NKVH + h) * DHEAD + d];
  red[d] = x * x; __syncthreads();
  for (int s = 64; s > 0; s >>= 1) { if (d < s) red[d] += red[d + s]; __syncthreads(); }
  float inv = rsqrtf(red[0] / DHEAD + 1e-6f);
  float v = wk[d] * (x * inv);
  xn[d] = v; __syncthreads();
  float rot = (d < 64) ? -xn[d + 64] : xn[d - 64];
  float out = v * cosE[t * DHEAD + d] + rot * sinE[t * DHEAD + d];
  long long off = ((long long)h * DHEAD + d) * SMAXC + (past + t);
  kOutL[off] = out;
  kc[off]    = out;
}

// ---------- V transpose into value cache [h][s][d] (out + aligned ws) ----------
__global__ __launch_bounds__(256)
void v_write_kernel(const float* __restrict__ v, float* __restrict__ vOutL,
                    float* __restrict__ vc, int past)
{
  int i = blockIdx.x * 256 + threadIdx.x;
  if (i >= TSEQ * NKVH * DHEAD) return;
  int d = i % DHEAD; int rest = i / DHEAD; int h = rest % NKVH; int t = rest / NKVH;
  long long off = ((long long)h * SMAXC + past + t) * DHEAD + d;
  float val = v[i];
  vOutL[off] = val;
  vc[off]    = val;
}

// ---------------------------------------------------------------------------
// Causal masked softmax. The score row (4KB) is DMA'd into LDS by the Tensor
// Data Mover (tensor_load_to_lds, TENSORcnt); the mask row arrives via
// per-lane async copies (global_load_async_to_lds_b128, ASYNCcnt).
// ---------------------------------------------------------------------------
__global__ __launch_bounds__(256)
void softmax_kernel(float* __restrict__ scores, const float* __restrict__ amask)
{
  __shared__ __align__(16) float srow[TSEQ];
  __shared__ __align__(16) float smask[TSEQ];
  __shared__ float red[256];
  const int h = blockIdx.x, t = blockIdx.y, tid = threadIdx.x;
  float* row = scores + ((long long)h * TSEQ + t) * TSEQ;
  const float* mrow = amask + (long long)t * TSEQ;

  // TDM: one descriptor moves the whole 1x1024 fp32 tile into LDS
  if (tid == 0) {
    unsigned long long g = (unsigned long long)row;
    unsigned int lds = (unsigned int)(unsigned long long)&srow[0];
    v4u g0;
    g0[0] = 1u;                                               // count=1 (valid user D#)
    g0[1] = lds;                                              // lds_addr
    g0[2] = (unsigned int)g;                                  // global_addr[31:0]
    g0[3] = (unsigned int)((g >> 32) & 0x1FFFFFFull) | (2u << 30); // addr[56:32] | type=2
    v8i g1;
    g1[0] = 0x20000;                                          // data_size=4B (code 2)
    g1[1] = (int)((TSEQ & 0xFFFF) << 16);                     // tensor_dim0 lo16
    g1[2] = (int)(((TSEQ >> 16) & 0xFFFF) | (1u << 16));      // tensor_dim0 hi | tensor_dim1 lo
    g1[3] = (int)((unsigned)(TSEQ & 0xFFFF) << 16);           // tensor_dim1 hi | tile_dim0
    g1[4] = 1;                                                // tile_dim1=1, tile_dim2=0
    g1[5] = TSEQ;                                             // tensor_dim0_stride lo32
    g1[6] = 0;
    g1[7] = 0;
    v4i zz = {0, 0, 0, 0};
    v8i z8 = {0, 0, 0, 0, 0, 0, 0, 0};
    __builtin_amdgcn_tensor_load_to_lds(g0, g1, zz, zz, z8, 0);
  }
  // async copy: each lane moves 16B of the mask row global -> LDS
  {
    const float* msrc = mrow + tid * 4;
    unsigned int mdst = (unsigned int)(unsigned long long)&smask[tid * 4];
    asm volatile("global_load_async_to_lds_b128 %0, %1, off"
                 :: "v"(mdst), "v"(msrc) : "memory");
    asm volatile("s_wait_asynccnt 0x0" ::: "memory");
  }
  __builtin_amdgcn_s_wait_tensorcnt(0);
  __syncthreads();

  float mx = -1e30f;
  for (int s = tid; s < TSEQ; s += 256) {
    float v = srow[s];
    if (s > t) v += -128.0f * smask[s];
    srow[s] = v;
    mx = fmaxf(mx, v);
  }
  __syncthreads();
  red[tid] = mx; __syncthreads();
  for (int s2 = 128; s2 > 0; s2 >>= 1) {
    if (tid < s2) red[tid] = fmaxf(red[tid], red[tid + s2]);
    __syncthreads();
  }
  mx = red[0]; __syncthreads();

  float sum = 0.f;
  for (int s = tid; s < TSEQ; s += 256) {
    float e = __expf(srow[s] - mx);
    srow[s] = e; sum += e;
  }
  __syncthreads();
  sum = block_reduce_sum_256(sum, red);
  float inv = 1.0f / sum;
  for (int s = tid; s < TSEQ; s += 256) row[s] = srow[s] * inv;
}

// ---------------- SwiGLU ----------------
__global__ __launch_bounds__(256)
void swiglu_kernel(const float* __restrict__ g, const float* __restrict__ u,
                   float* __restrict__ a, int n)
{
  int i = blockIdx.x * 256 + threadIdx.x;
  if (i < n) { float x = g[i]; a[i] = (x / (1.f + __expf(-x))) * u[i]; }
}

// ---------------- final RMSNorm of last token ----------------
__global__ __launch_bounds__(256)
void final_norm_kernel(const float* __restrict__ hidden, const float* __restrict__ w,
                       float* __restrict__ lastn)
{
  __shared__ float red[256];
  const float* x = hidden + (long long)(TSEQ - 1) * DMODEL;
  float ss = 0.f;
  for (int c = threadIdx.x; c < DMODEL; c += 256) { float v = x[c]; ss += v * v; }
  ss = block_reduce_sum_256(ss, red);
  float inv = rsqrtf(ss / DMODEL + 1e-6f);
  for (int c = threadIdx.x; c < DMODEL; c += 256) lastn[c] = w[c] * (x[c] * inv);
}

// ---------------- LM head GEMV + per-block argmax ----------------
__global__ __launch_bounds__(256)
void lm_kernel(const float* __restrict__ lastn, const float* __restrict__ Wlm,
               float* __restrict__ pv, int* __restrict__ pi)
{
  __shared__ float rv[256]; __shared__ int ri[256];
  int col = blockIdx.x * 256 + threadIdx.x;
  float dot = -1e30f;
  if (col < VOCAB) {
    float s = 0.f;
    for (int d = 0; d < DMODEL; ++d) s += lastn[d] * Wlm[(long long)d * VOCAB + col];
    dot = s;
  }
  rv[threadIdx.x] = dot; ri[threadIdx.x] = col;
  __syncthreads();
  for (int s2 = 128; s2 > 0; s2 >>= 1) {
    if (threadIdx.x < s2) {
      float ov = rv[threadIdx.x + s2]; int oi = ri[threadIdx.x + s2];
      if (ov > rv[threadIdx.x] || (ov == rv[threadIdx.x] && oi < ri[threadIdx.x])) {
        rv[threadIdx.x] = ov; ri[threadIdx.x] = oi;
      }
    }
    __syncthreads();
  }
  if (threadIdx.x == 0) { pv[blockIdx.x] = rv[0]; pi[blockIdx.x] = ri[0]; }
}

__global__ __launch_bounds__(256)
void argmax_kernel(const float* __restrict__ pv, const int* __restrict__ pi, int nb,
                   float* __restrict__ dout)
{
  __shared__ float rv[256]; __shared__ int ri[256];
  float v = -1e30f; int idx = 0x7fffffff;
  for (int b = threadIdx.x; b < nb; b += 256) {
    float ov = pv[b]; int oi = pi[b];
    if (ov > v || (ov == v && oi < idx)) { v = ov; idx = oi; }
  }
  rv[threadIdx.x] = v; ri[threadIdx.x] = idx; __syncthreads();
  for (int s2 = 128; s2 > 0; s2 >>= 1) {
    if (threadIdx.x < s2) {
      float ov = rv[threadIdx.x + s2]; int oi = ri[threadIdx.x + s2];
      if (ov > rv[threadIdx.x] || (ov == rv[threadIdx.x] && oi < ri[threadIdx.x])) {
        rv[threadIdx.x] = ov; ri[threadIdx.x] = oi;
      }
    }
    __syncthreads();
  }
  if (threadIdx.x == 0) {
    ((int*)dout)[0] = ri[0];                                      // tok_id
    size_t kvN = (size_t)LAYERS * NKVH * DHEAD * SMAXC;
    ((int*)dout)[1 + 2 * kvN] = TSEQ;                             // new_seq_len (past=0)
  }
}

__global__ __launch_bounds__(256)
void zero_kernel(float* __restrict__ p, long long n)
{
  long long i = (long long)blockIdx.x * 256 + threadIdx.x;
  long long stride = (long long)gridDim.x * 256;
  for (; i < n; i += stride) p[i] = 0.f;
}

// ---------------------------------------------------------------------------
extern "C" void kernel_launch(void* const* d_in, const int* in_sizes, int n_in,
                              void* d_out, int out_size, void* d_ws, size_t ws_size,
                              hipStream_t stream)
{
  (void)in_sizes; (void)n_in; (void)out_size; (void)ws_size;
  const int*   input_ids = (const int*)d_in[2];
  const float* amask     = (const float*)d_in[3];
  const unsigned char* edata = (const unsigned char*)d_in[5];
  const float* escale = (const float*)d_in[6];
  const float* ezero  = (const float*)d_in[7];
  const float* cosE   = (const float*)d_in[8];
  const float* sinE   = (const float*)d_in[9];
  const float* w_in_ln= (const float*)d_in[10];
  const float* Wq     = (const float*)d_in[11];
  const float* Wk     = (const float*)d_in[12];
  const float* Wv     = (const float*)d_in[13];
  const float* wqn    = (const float*)d_in[14];
  const float* wkn    = (const float*)d_in[15];
  const float* Wo     = (const float*)d_in[16];
  const float* w_post = (const float*)d_in[17];
  const float* Wg     = (const float*)d_in[18];
  const float* Wu     = (const float*)d_in[19];
  const float* Wd     = (const float*)d_in[20];
  const float* wfin   = (const float*)d_in[21];
  const float* Wlm    = (const float*)d_in[22];

  float* out = (float*)d_out;
  const long long KVN = (long long)LAYERS * NKVH * DHEAD * SMAXC;
  float* kOut = out + 1;          // key_buffer output (4B-misaligned base)
  float* vOut = out + 1 + KVN;    // value_buffer output

  // workspace carve-out (fp32, 16B-aligned chunks)
  float* ws = (float*)d_ws;
  float* hidden = ws; ws += (long long)TSEQ * DMODEL;
  float* hn     = ws; ws += (long long)TSEQ * DMODEL;
  float* qbuf   = ws; ws += (long long)TSEQ * NH * DHEAD;
  float* kbuf   = ws; ws += (long long)TSEQ * NKVH * DHEAD;
  float* vbuf   = ws; ws += (long long)TSEQ * NKVH * DHEAD;
  float* qrot   = ws; ws += (long long)NH * TSEQ * DHEAD;
  float* scores = ws; ws += (long long)NH * TSEQ * TSEQ;
  float* ctx    = ws; ws += (long long)TSEQ * NH * DHEAD;
  float* gate   = ws; ws += (long long)TSEQ * FF;
  float* up     = ws; ws += (long long)TSEQ * FF;
  float* act    = ws; ws += (long long)TSEQ * FF;
  float* kc     = ws; ws += (long long)NKVH * DHEAD * SMAXC;  // aligned K-cache mirror
  float* vc     = ws; ws += (long long)NKVH * SMAXC * DHEAD;  // aligned V-cache mirror
  float* lastn  = ws; ws += DMODEL;
  float* pv     = ws; ws += 256;
  int*   pi     = (int*)ws;

  zero_kernel<<<1024, 256, 0, stream>>>(kOut, 2 * KVN);
  embed_kernel<<<(TSEQ * DMODEL + 255) / 256, 256, 0, stream>>>(
      input_ids, edata, escale, ezero, hidden);

  auto gemm = [&](const float* A, int lda, long long sA,
                  const float* B, int ldb, long long sB, int zdivB,
                  float* C, int ldc, long long sC,
                  int M, int N, int K, int batch, int addC) {
    dim3 grid(N / 128, M / 128, batch);
    gemm_bf16_wmma<<<grid, 256, 0, stream>>>(A, lda, sA, B, ldb, sB, zdivB,
                                             C, ldc, sC, M, N, K, addC);
  };

  for (int l = 0; l < LAYERS; ++l) {
    rmsnorm_kernel<<<TSEQ, 256, 0, stream>>>(hidden, hn, w_in_ln + l * DMODEL, DMODEL);

    gemm(hn, DMODEL, 0, Wq + (long long)l * DMODEL * NH * DHEAD, NH * DHEAD, 0, 0,
         qbuf, NH * DHEAD, 0, TSEQ, NH * DHEAD, DMODEL, 1, 0);
    gemm(hn, DMODEL, 0, Wk + (long long)l * DMODEL * NKVH * DHEAD, NKVH * DHEAD, 0, 0,
         kbuf, NKVH * DHEAD, 0, TSEQ, NKVH * DHEAD, DMODEL, 1, 0);
    gemm(hn, DMODEL, 0, Wv + (long long)l * DMODEL * NKVH * DHEAD, NKVH * DHEAD, 0, 0,
         vbuf, NKVH * DHEAD, 0, TSEQ, NKVH * DHEAD, DMODEL, 1, 0);

    float* kOutL = kOut + (long long)l * NKVH * DHEAD * SMAXC;
    float* vOutL = vOut + (long long)l * NKVH * SMAXC * DHEAD;
    q_rope_kernel<<<dim3(NH, TSEQ), 128, 0, stream>>>(qbuf, qrot, wqn + l * DHEAD, cosE, sinE);
    k_rope_kernel<<<dim3(NKVH, TSEQ), 128, 0, stream>>>(kbuf, kOutL, kc,
                                                        wkn + l * DHEAD, cosE, sinE, 0);
    v_write_kernel<<<(TSEQ * NKVH * DHEAD + 255) / 256, 256, 0, stream>>>(vbuf, vOutL, vc, 0);

    // scores[h] = qrot[h] (T x DH) @ Kcache[h>>1] (DH x S)   (cache is K^T already)
    gemm(qrot, DHEAD, (long long)TSEQ * DHEAD,
         kc, SMAXC, (long long)DHEAD * SMAXC, 1,
         scores, TSEQ, (long long)TSEQ * TSEQ,
         TSEQ, TSEQ, DHEAD, NH, 0);

    softmax_kernel<<<dim3(NH, TSEQ), 256, 0, stream>>>(scores, amask);

    // ctx[:, h*DH:(h+1)*DH] = attn[h] (T x T) @ Vcache[h>>1] (T x DH)
    gemm(scores, TSEQ, (long long)TSEQ * TSEQ,
         vc, DHEAD, (long long)SMAXC * DHEAD, 1,
         ctx, NH * DHEAD, (long long)DHEAD,
         TSEQ, DHEAD, TSEQ, NH, 0);

    // hidden += ctx @ Wo
    gemm(ctx, NH * DHEAD, 0, Wo + (long long)l * NH * DHEAD * DMODEL, DMODEL, 0, 0,
         hidden, DMODEL, 0, TSEQ, DMODEL, NH * DHEAD, 1, 1);

    rmsnorm_kernel<<<TSEQ, 256, 0, stream>>>(hidden, hn, w_post + l * DMODEL, DMODEL);
    gemm(hn, DMODEL, 0, Wg + (long long)l * DMODEL * FF, FF, 0, 0,
         gate, FF, 0, TSEQ, FF, DMODEL, 1, 0);
    gemm(hn, DMODEL, 0, Wu + (long long)l * DMODEL * FF, FF, 0, 0,
         up, FF, 0, TSEQ, FF, DMODEL, 1, 0);
    swiglu_kernel<<<(TSEQ * FF + 255) / 256, 256, 0, stream>>>(gate, up, act, TSEQ * FF);
    // hidden += act @ Wdown
    gemm(act, FF, 0, Wd + (long long)l * FF * DMODEL, DMODEL, 0, 0,
         hidden, DMODEL, 0, TSEQ, DMODEL, FF, 1, 1);
  }

  final_norm_kernel<<<1, 256, 0, stream>>>(hidden, wfin, lastn);
  lm_kernel<<<(VOCAB + 255) / 256, 256, 0, stream>>>(lastn, Wlm, pv, pi);
  argmax_kernel<<<1, 256, 0, stream>>>(pv, pi, (VOCAB + 255) / 256, out);
}
